// CausalSelfAttention_69561290326594
// MI455X (gfx1250) — matrix operand order repaired and checked by reading, not compiled, not run on hardware
//
#include <hip/hip_runtime.h>
#include <hip/hip_bf16.h>
#include <math.h>

// CausalSelfAttention forward for MI455X (gfx1250), bf16 WMMA pipeline.
// B=2, T=2048, C=1024, H=16, D=64.
//
// Stage 0: fp32 -> bf16 conversion pre-pass for x and all weights (once).
// Stage 1: Q/K/V projections, pure-bf16 WMMA GEMM, 32x32 tile per wave.
//          (layout templated: Q/K row-major [B,H,T,D], V transposed [B,H,D,T]
//           with packed 16B stores)
// Stage 2: flash attention, 16 query rows per wave, online softmax.
// Stage 3: output projection, bf16 WMMA GEMM, fp32 stores.

typedef __attribute__((ext_vector_type(16))) __bf16 v16bf;
typedef __attribute__((ext_vector_type(8)))  __bf16 v8bf;
typedef __attribute__((ext_vector_type(8)))  float  v8f;

static __device__ __forceinline__ __bf16 to_bf16(float f) {
  unsigned u = __builtin_bit_cast(unsigned, f);
  unsigned rr = u + 0x7FFFu + ((u >> 16) & 1u);   // round-to-nearest-even
  unsigned short hs = (unsigned short)(rr >> 16);
  return __builtin_bit_cast(__bf16, hs);
}

static __device__ __forceinline__ v16bf cat8(v8bf lo, v8bf hi) {
  return __builtin_shufflevector(lo, hi, 0,1,2,3,4,5,6,7,8,9,10,11,12,13,14,15);
}

static __device__ __forceinline__ v8f wmma_bf16(v16bf a, v16bf b, v8f c) {
  // (neg_a, A, neg_b, B, c_mod, C, reuse_a, reuse_b)
  return __builtin_amdgcn_wmma_f32_16x16x32_bf16(false, a, false, b, (short)0, c,
                                                 false, false);
}

// A fragment (16x32 bf16): lane holds row r; elems 0..15 = K+hl*16 .. K+hl*16+15
static __device__ __forceinline__ v16bf load_a_frag(const __bf16* rowp, int kk, int hl) {
  const __bf16* p = rowp + kk + hl * 16;
  return cat8(*(const v8bf*)p, *(const v8bf*)(p + 8));
}

// ---------------------------------------------------------------------------
// Stage 0: fp32 -> bf16 bulk convert (n multiple of 8)
// ---------------------------------------------------------------------------
__global__ __launch_bounds__(256) void cvt_bf16_kernel(
    const float* __restrict__ in, __bf16* __restrict__ out, int n) {
  int i = (blockIdx.x * 256 + threadIdx.x) * 8;
  if (i + 8 > n) return;
  float4 f0 = *(const float4*)(in + i);
  float4 f1 = *(const float4*)(in + i + 4);
  v8bf o;
  o[0] = to_bf16(f0.x); o[1] = to_bf16(f0.y);
  o[2] = to_bf16(f0.z); o[3] = to_bf16(f0.w);
  o[4] = to_bf16(f1.x); o[5] = to_bf16(f1.y);
  o[6] = to_bf16(f1.z); o[7] = to_bf16(f1.w);
  *(v8bf*)(out + i) = o;
}

// ---------------------------------------------------------------------------
// Stage 1: y = x @ W^T  (M=4096, N=1024, K=1024), all bf16, WMMA.
// Wave tile 32x32 (2x2 WMMA sub-tiles). Block = 8 waves as 2(M) x 4(N) ->
// block tile 64x128. Grid = (N/128, M/64) = (8, 64).
// LAYOUT 0: out[((b*16+h)*2048+t)*64+d]   ([B,H,T,D])   for Q, K
// LAYOUT 1: out[((b*16+h)*64+d)*2048+t]   ([B,H,D,T])   for V (transposed);
//           8 acc elems of a sub-tile are consecutive in t -> one 16B store.
// ---------------------------------------------------------------------------
template <int LAYOUT>
__global__ __launch_bounds__(256) void qkv_proj_kernel(
    const __bf16* __restrict__ X, const __bf16* __restrict__ W,
    __bf16* __restrict__ Out) {
  const int tid  = threadIdx.x;
  const int lane = tid & 31;
  const int wave = tid >> 5;
  const int r  = lane & 15;
  const int hl = lane >> 4;
  const int row0 = blockIdx.y * 64 + (wave >> 2) * 32;
  const int col0 = blockIdx.x * 128 + (wave & 3) * 32;

  const __bf16* ar0 = X + (size_t)(row0 + r) * 1024;
  const __bf16* ar1 = X + (size_t)(row0 + 16 + r) * 1024;
  const __bf16* bc0 = W + (size_t)(col0 + r) * 1024;
  const __bf16* bc1 = W + (size_t)(col0 + 16 + r) * 1024;

  v8f acc00 = {}, acc01 = {}, acc10 = {}, acc11 = {};

  for (int kk = 0; kk < 1024; kk += 32) {
    if (kk + 128 < 1024) {               // cover L2->L0 latency
      __builtin_prefetch(ar0 + kk + 128, 0, 3);
      __builtin_prefetch(bc0 + kk + 128, 0, 3);
    }
    v16bf a0 = load_a_frag(ar0, kk, hl);
    v16bf a1 = load_a_frag(ar1, kk, hl);
    v16bf b0 = *(const v16bf*)(bc0 + kk + hl * 16);
    v16bf b1 = *(const v16bf*)(bc1 + kk + hl * 16);
    acc00 = wmma_bf16(a0, b0, acc00);
    acc01 = wmma_bf16(a0, b1, acc01);
    acc10 = wmma_bf16(a1, b0, acc10);
    acc11 = wmma_bf16(a1, b1, acc11);
  }

  const v8f accs[2][2] = {{acc00, acc01}, {acc10, acc11}};
#pragma unroll
  for (int sm = 0; sm < 2; ++sm) {
#pragma unroll
    for (int sn = 0; sn < 2; ++sn) {
      const v8f acc = accs[sm][sn];
      if (LAYOUT == 0) {
#pragma unroll
        for (int i = 0; i < 8; ++i) {         // C/D: m = i + 8*hl, n = r
          int m = row0 + sm * 16 + i + 8 * hl;     // b*T + t
          int n = col0 + sn * 16 + r;              // h*D + d
          int bb = m >> 11, tt = m & 2047;
          int hh = n >> 6,  dd = n & 63;
          Out[((size_t)(bb * 16 + hh) * 2048 + tt) * 64 + dd] = to_bf16(acc[i]);
        }
      } else {
        // acc[i] for i=0..7 -> t = t0 + i at fixed d: one contiguous 16B store
        int m0 = row0 + sm * 16 + 8 * hl;          // multiple of 8
        int n  = col0 + sn * 16 + r;
        int bb = m0 >> 11, t0 = m0 & 2047;
        int hh = n >> 6,   dd = n & 63;
        v8bf o;
#pragma unroll
        for (int i = 0; i < 8; ++i) o[i] = to_bf16(acc[i]);
        *(v8bf*)(Out + ((size_t)(bb * 16 + hh) * 64 + dd) * 2048 + t0) = o;
      }
    }
  }
}

// ---------------------------------------------------------------------------
// Stage 2: flash attention. One wave = 16 query rows of one (b,h).
// Q,K: [BH,2048,64] bf16; Vt: [BH,64,2048] bf16; Y: [B,T,C] bf16.
// ---------------------------------------------------------------------------
__global__ __launch_bounds__(256) void attn_kernel(
    const __bf16* __restrict__ Q, const __bf16* __restrict__ K,
    const __bf16* __restrict__ Vt, __bf16* __restrict__ Y) {
  __shared__ __align__(32) __bf16 lds[8][16 * 40];   // stride 40: no bank conflicts

  const int tid  = threadIdx.x;
  const int lane = tid & 31;
  const int wave = tid >> 5;
  const int r  = lane & 15;
  const int hl = lane >> 4;
  const int wglob = blockIdx.x * 8 + wave;
  const int bh = wglob >> 7;            // 128 query blocks per (b,h)
  const int q0 = (wglob & 127) * 16;

  const __bf16* Qp = Q  + (size_t)bh * 2048 * 64;
  const __bf16* Kp = K  + (size_t)bh * 2048 * 64;
  const __bf16* Vp = Vt + (size_t)bh * 64 * 2048;

  v16bf qa[2];
#pragma unroll
  for (int c = 0; c < 2; ++c)
    qa[c] = load_a_frag(Qp + (size_t)(q0 + r) * 64, c * 32, hl);

  v8f o0 = {}, o1 = {}, o2 = {}, o3 = {};
  float mrow[8], lrow[8];
#pragma unroll
  for (int i = 0; i < 8; ++i) { mrow[i] = -__builtin_inff(); lrow[i] = 0.f; }

  const int kv_end = q0 + 16;           // keys 0 .. q0+15 needed
  for (int s0 = 0; s0 < kv_end; s0 += 32) {
    // S = Q @ K^T for 32 keys (two 16x16 tiles), reduction over D=64
    v8f st0 = {}, st1 = {};
#pragma unroll
    for (int c = 0; c < 2; ++c) {
      v16bf kb0 = *(const v16bf*)(Kp + (size_t)(s0 +      r) * 64 + c * 32 + hl * 16);
      v16bf kb1 = *(const v16bf*)(Kp + (size_t)(s0 + 16 + r) * 64 + c * 32 + hl * 16);
      st0 = wmma_bf16(qa[c], kb0, st0);
      st1 = wmma_bf16(qa[c], kb1, st1);
    }

    float sv0[8], sv1[8];
#pragma unroll
    for (int i = 0; i < 8; ++i) { sv0[i] = st0[i] * 0.125f; sv1[i] = st1[i] * 0.125f; }

    if (s0 + 31 > q0) {                 // causal mask on diagonal blocks
      const int kj0 = s0 + r, kj1 = s0 + 16 + r;
#pragma unroll
      for (int i = 0; i < 8; ++i) {
        const int qi = q0 + i + 8 * hl;
        if (kj0 > qi) sv0[i] = -__builtin_inff();
        if (kj1 > qi) sv1[i] = -__builtin_inff();
      }
    }

    float alpha[8];
#pragma unroll
    for (int i = 0; i < 8; ++i) {       // row max across 16-lane half
      float v = fmaxf(sv0[i], sv1[i]);
#pragma unroll
      for (int xm = 1; xm < 16; xm <<= 1) v = fmaxf(v, __shfl_xor(v, xm, 32));
      float mn = fmaxf(mrow[i], v);
      alpha[i] = __expf(mrow[i] - mn);
      mrow[i] = mn;
    }

#pragma unroll
    for (int i = 0; i < 8; ++i) {
      float p0 = __expf(sv0[i] - mrow[i]);
      float p1 = __expf(sv1[i] - mrow[i]);
      lds[wave][(i + 8 * hl) * 40 + r]      = to_bf16(p0);   // P[m][k] row-major
      lds[wave][(i + 8 * hl) * 40 + 16 + r] = to_bf16(p1);
      float t = p0 + p1;
#pragma unroll
      for (int xm = 1; xm < 16; xm <<= 1) t += __shfl_xor(t, xm, 32);
      lrow[i] = lrow[i] * alpha[i] + t;
      o0[i] *= alpha[i]; o1[i] *= alpha[i]; o2[i] *= alpha[i]; o3[i] *= alpha[i];
    }

    asm volatile("s_wait_dscnt 0" ::: "memory");  // LDS store -> load hazard

    // re-load P as A fragment (16x32)
    const __bf16* pl = &lds[wave][r * 40 + hl * 16];
    v16bf pa = cat8(*(const v8bf*)pl, *(const v8bf*)(pl + 8));

    // O += P @ V : B operand from transposed V, contiguous along keys
    o0 = wmma_bf16(pa, *(const v16bf*)(Vp + (size_t)( 0 + r) * 2048 + s0 + hl * 16), o0);
    o1 = wmma_bf16(pa, *(const v16bf*)(Vp + (size_t)(16 + r) * 2048 + s0 + hl * 16), o1);
    o2 = wmma_bf16(pa, *(const v16bf*)(Vp + (size_t)(32 + r) * 2048 + s0 + hl * 16), o2);
    o3 = wmma_bf16(pa, *(const v16bf*)(Vp + (size_t)(48 + r) * 2048 + s0 + hl * 16), o3);
  }

  // finalize: divide by row sums, write to [B,T,C] bf16 workspace
  const int bb = bh >> 4, hh = bh & 15;
#pragma unroll
  for (int i = 0; i < 8; ++i) {
    const float inv = 1.0f / lrow[i];
    const size_t base =
        (size_t)(bb * 2048 + q0 + i + 8 * hl) * 1024 + hh * 64 + r;
    Y[base +  0] = to_bf16(o0[i] * inv);
    Y[base + 16] = to_bf16(o1[i] * inv);
    Y[base + 32] = to_bf16(o2[i] * inv);
    Y[base + 48] = to_bf16(o3[i] * inv);
  }
}

// ---------------------------------------------------------------------------
// Stage 3: out = y @ Wp^T  (bf16 A and B, fp32 out), 32x32 tile per wave.
// ---------------------------------------------------------------------------
__global__ __launch_bounds__(256) void out_proj_kernel(
    const __bf16* __restrict__ Yin, const __bf16* __restrict__ W,
    float* __restrict__ Out) {
  const int tid  = threadIdx.x;
  const int lane = tid & 31;
  const int wave = tid >> 5;
  const int r  = lane & 15;
  const int hl = lane >> 4;
  const int row0 = blockIdx.y * 64 + (wave >> 2) * 32;
  const int col0 = blockIdx.x * 128 + (wave & 3) * 32;

  const __bf16* ar0 = Yin + (size_t)(row0 + r) * 1024;
  const __bf16* ar1 = Yin + (size_t)(row0 + 16 + r) * 1024;
  const __bf16* bc0 = W + (size_t)(col0 + r) * 1024;
  const __bf16* bc1 = W + (size_t)(col0 + 16 + r) * 1024;

  v8f acc00 = {}, acc01 = {}, acc10 = {}, acc11 = {};

  for (int kk = 0; kk < 1024; kk += 32) {
    if (kk + 128 < 1024) {
      __builtin_prefetch(ar0 + kk + 128, 0, 3);
      __builtin_prefetch(bc0 + kk + 128, 0, 3);
    }
    v16bf a0 = load_a_frag(ar0, kk, hl);
    v16bf a1 = load_a_frag(ar1, kk, hl);
    v16bf b0 = *(const v16bf*)(bc0 + kk + hl * 16);
    v16bf b1 = *(const v16bf*)(bc1 + kk + hl * 16);
    acc00 = wmma_bf16(a0, b0, acc00);
    acc01 = wmma_bf16(a0, b1, acc01);
    acc10 = wmma_bf16(a1, b0, acc10);
    acc11 = wmma_bf16(a1, b1, acc11);
  }

#pragma unroll
  for (int i = 0; i < 8; ++i) {
    Out[(size_t)(row0 +      i + 8 * hl) * 1024 + col0 +      r] = acc00[i];
    Out[(size_t)(row0 +      i + 8 * hl) * 1024 + col0 + 16 + r] = acc01[i];
    Out[(size_t)(row0 + 16 + i + 8 * hl) * 1024 + col0 +      r] = acc10[i];
    Out[(size_t)(row0 + 16 + i + 8 * hl) * 1024 + col0 + 16 + r] = acc11[i];
  }
}

// ---------------------------------------------------------------------------
extern "C" void kernel_launch(void* const* d_in, const int* in_sizes, int n_in,
                              void* d_out, int out_size, void* d_ws, size_t ws_size,
                              hipStream_t stream) {
  (void)in_sizes; (void)n_in; (void)out_size;
  const float* x  = (const float*)d_in[0];
  const float* Wk = (const float*)d_in[1];
  const float* Wq = (const float*)d_in[2];
  const float* Wv = (const float*)d_in[3];
  const float* Wp = (const float*)d_in[4];

  const size_t XE = (size_t)2 * 2048 * 1024;   // 4,194,304 (x / per-tensor qkvy)
  const size_t WE = (size_t)1024 * 1024;       // 1,048,576 (weights)
  const size_t TOTAL = XE + 4 * WE + 4 * XE;   // bf16 elements
  if (ws_size < TOTAL * sizeof(__bf16)) return;  // deterministic guard

  __bf16* xb  = (__bf16*)d_ws;
  __bf16* wqb = xb  + XE;
  __bf16* wkb = wqb + WE;
  __bf16* wvb = wkb + WE;
  __bf16* wpb = wvb + WE;
  __bf16* q_ws  = wpb + WE;
  __bf16* k_ws  = q_ws  + XE;
  __bf16* vt_ws = k_ws  + XE;
  __bf16* y_ws  = vt_ws + XE;

  dim3 block(256);

  // Stage 0: bulk fp32 -> bf16
  cvt_bf16_kernel<<<(unsigned)(XE / 2048), block, 0, stream>>>(x,  xb,  (int)XE);
  cvt_bf16_kernel<<<(unsigned)(WE / 2048), block, 0, stream>>>(Wq, wqb, (int)WE);
  cvt_bf16_kernel<<<(unsigned)(WE / 2048), block, 0, stream>>>(Wk, wkb, (int)WE);
  cvt_bf16_kernel<<<(unsigned)(WE / 2048), block, 0, stream>>>(Wv, wvb, (int)WE);
  cvt_bf16_kernel<<<(unsigned)(WE / 2048), block, 0, stream>>>(Wp, wpb, (int)WE);

  // Stage 1: projections (block tile 64x128, wave tile 32x32)
  dim3 gproj(8, 64);
  qkv_proj_kernel<0><<<gproj, block, 0, stream>>>(xb, wqb, q_ws);
  qkv_proj_kernel<0><<<gproj, block, 0, stream>>>(xb, wkb, k_ws);
  qkv_proj_kernel<1><<<gproj, block, 0, stream>>>(xb, wvb, vt_ws);  // V transposed

  // Stage 2: attention (4096 waves, 16 query rows each)
  attn_kernel<<<512, block, 0, stream>>>(q_ws, k_ws, vt_ws, y_ws);

  // Stage 3: output projection
  out_proj_kernel<<<gproj, block, 0, stream>>>(y_ws, wpb, (float*)d_out);
}
